// LinearEncoder_61718680044349
// MI455X (gfx1250) — matrix operand order repaired and checked by reading, not compiled, not run on hardware
//
#include <hip/hip_runtime.h>

#define N_NODES 100000
#define N_EDGES 1600000
#define IN_C    512
#define OUT_C   128

typedef __attribute__((ext_vector_type(16))) __bf16 v16bf;
typedef __attribute__((ext_vector_type(2)))  __bf16 v2bf;
typedef __attribute__((ext_vector_type(8)))  float  v8f;
typedef __attribute__((ext_vector_type(2)))  float  v2f;

// ---------------- degree / normalization ----------------

__global__ void __launch_bounds__(256) gcn_init_deg(float* deg) {
    int i = blockIdx.x * 256 + threadIdx.x;
    if (i < N_NODES) deg[i] = 1.0f;                  // self-loop
}

__global__ void __launch_bounds__(256) gcn_count_deg(const int* __restrict__ ei, float* deg) {
    int e = blockIdx.x * 256 + threadIdx.x;
    if (e < N_EDGES) atomicAdd(&deg[ei[N_EDGES + e]], 1.0f);
}

__global__ void __launch_bounds__(256) gcn_rsqrt(float* deg) {
    int i = blockIdx.x * 256 + threadIdx.x;
    if (i < N_NODES) deg[i] = rsqrtf(deg[i]);        // in place: deg -> dinv
}

// ---------------- h = x @ W  (bf16x3 WMMA, f32 accumulate) ----------------

__global__ void __launch_bounds__(256) gcn_gemm(const float* __restrict__ x,
                                                const float* __restrict__ W,
                                                float* __restrict__ h) {
    extern __shared__ __bf16 smem[];
    __bf16* whi = smem;                              // [OUT_C][IN_C] transposed, bf16 hi
    __bf16* wlo = smem + OUT_C * IN_C;               // [OUT_C][IN_C] transposed, bf16 lo

    const int tid = threadIdx.x;
    // cooperative convert + transpose of W (k-major in memory -> n-major in LDS)
    for (int idx = tid; idx < IN_C * OUT_C; idx += 256) {
        const int k = idx >> 7;                      // 0..511
        const int n = idx & (OUT_C - 1);             // 0..127
        float w  = W[idx];
        __bf16 hi = (__bf16)w;                       // native RNE f32->bf16
        __bf16 lo = (__bf16)(w - (float)hi);
        whi[n * IN_C + k] = hi;
        wlo[n * IN_C + k] = lo;
    }
    __syncthreads();

    const int wave = tid >> 5;
    const int lane = tid & 31;
    const int rowBase = (blockIdx.x * 8 + wave) * 16;
    if (rowBase >= N_NODES) return;                  // wave-uniform: EXEC all-1 for WMMA

    const int half = lane >> 4;                      // 0 | 1
    const int mrow = lane & 15;
    const float* xrow = x + (long)(rowBase + mrow) * IN_C;

    union Frag { uint4 q[2]; v16bf v; };
    union A16  { v2bf p[8]; v16bf v; };              // packed bf16x2 per VGPR

    v8f acc[8];
#pragma unroll
    for (int t = 0; t < 8; ++t) acc[t] = (v8f){0.f,0.f,0.f,0.f,0.f,0.f,0.f,0.f};

    for (int k0 = 0; k0 < IN_C; k0 += 32) {
        // ---- A fragment (16-bit A 16x32 lane layout) ----
        const int ka = k0 + half * 8;                // lanes 0-15: K 0..7/16..23 ; 16-31: 8..15/24..31
        float4 f0 = *(const float4*)(xrow + ka);
        float4 f1 = *(const float4*)(xrow + ka + 4);
        float4 f2 = *(const float4*)(xrow + ka + 16);
        float4 f3 = *(const float4*)(xrow + ka + 20);
        v2f fp[8] = { {f0.x,f0.y}, {f0.z,f0.w}, {f1.x,f1.y}, {f1.z,f1.w},
                      {f2.x,f2.y}, {f2.z,f2.w}, {f3.x,f3.y}, {f3.z,f3.w} };
        A16 ah, al;
#pragma unroll
        for (int p = 0; p < 8; ++p) {
            v2bf hb = __builtin_convertvector(fp[p], v2bf);        // v_cvt_pk_bf16_f32
            v2f  hw = __builtin_convertvector(hb, v2f);            // widen (shl/mask)
            al.p[p] = __builtin_convertvector(fp[p] - hw, v2bf);   // v_cvt_pk_bf16_f32
            ah.p[p] = hb;
        }

        // ---- 8 N-tiles, B fragments from LDS (lanes0-15: K 0..7/16..23, lanes16-31: K 8..15/24..31) ----
        const int kb = k0 + half * 8;
#pragma unroll
        for (int t = 0; t < 8; ++t) {
            const __bf16* bh = whi + (t * 16 + mrow) * IN_C + kb;
            const __bf16* bl = wlo + (t * 16 + mrow) * IN_C + kb;
            Frag bhi, blo;
            bhi.q[0] = *(const uint4*)(bh);
            bhi.q[1] = *(const uint4*)(bh + 16);
            blo.q[0] = *(const uint4*)(bl);
            blo.q[1] = *(const uint4*)(bl + 16);
            acc[t] = __builtin_amdgcn_wmma_f32_16x16x32_bf16(false, ah.v, false, bhi.v,
                                                             (short)0, acc[t], false, false);
            acc[t] = __builtin_amdgcn_wmma_f32_16x16x32_bf16(false, ah.v, false, blo.v,
                                                             (short)0, acc[t], false, false);
            acc[t] = __builtin_amdgcn_wmma_f32_16x16x32_bf16(false, al.v, false, bhi.v,
                                                             (short)0, acc[t], false, false);
        }
    }

    // ---- store C/D: VGPR v -> row (v + 8*half), col = t*16 + mrow ----
    float* hrow = h + (long)rowBase * OUT_C;
#pragma unroll
    for (int t = 0; t < 8; ++t) {
#pragma unroll
        for (int v = 0; v < 8; ++v) {
            hrow[(v + 8 * half) * OUT_C + t * 16 + mrow] = acc[t][v];
        }
    }
}

// ---------------- out init: self-loop + bias (and tail zero) ----------------

__global__ void __launch_bounds__(256) gcn_init_out(const float* __restrict__ h,
                                                    const float* __restrict__ dinv,
                                                    const float* __restrict__ b,
                                                    float* out, int out_size) {
    int idx = blockIdx.x * 256 + threadIdx.x;
    if (idx >= out_size) return;
    if (idx < N_NODES * OUT_C) {
        int i = idx >> 7;
        float di = dinv[i];
        out[idx] = h[idx] * di * di + b[idx & (OUT_C - 1)];
    } else {
        out[idx] = 0.0f;                             // tuple's scalar 0
    }
}

// ---------------- edge scatter: one wave per edge, 4 floats per lane ----------------
// e is forced wave-uniform (readfirstlane) so the edge/degree loads scalarize;
// h (51.2 MB) and out (51.2 MB) are L2-resident (192 MB), so the gather +
// 205M global_atomic_add_f32 run at L2 rate.

__global__ void __launch_bounds__(256) gcn_scatter(const int* __restrict__ ei,
                                                   const float* __restrict__ h,
                                                   const float* __restrict__ dinv,
                                                   float* out) {
    const int lane = threadIdx.x & 31;
    const int e = __builtin_amdgcn_readfirstlane(
        (int)(blockIdx.x * 8u + (threadIdx.x >> 5)));
    if (e >= N_EDGES) return;                        // uniform branch
    const int src = __builtin_amdgcn_readfirstlane(ei[e]);
    const int dst = __builtin_amdgcn_readfirstlane(ei[N_EDGES + e]);
    const float norm = dinv[src] * dinv[dst];
    const float4 v = *(const float4*)(h + (size_t)src * OUT_C + lane * 4);
    float* o = out + (size_t)dst * OUT_C + lane * 4;
    atomicAdd(o + 0, v.x * norm);
    atomicAdd(o + 1, v.y * norm);
    atomicAdd(o + 2, v.z * norm);
    atomicAdd(o + 3, v.w * norm);
}

// ---------------- launcher ----------------

extern "C" void kernel_launch(void* const* d_in, const int* in_sizes, int n_in,
                              void* d_out, int out_size, void* d_ws, size_t ws_size,
                              hipStream_t stream) {
    const float* x  = (const float*)d_in[0];
    const int*   ei = (const int*)d_in[1];
    const float* W  = (const float*)d_in[2];
    const float* b  = (const float*)d_in[3];
    float* out  = (float*)d_out;

    float* hbuf = (float*)d_ws;                                // [N, OUT_C] f32 (51.2 MB)
    float* deg  = hbuf + (size_t)N_NODES * OUT_C;              // [N] f32 (becomes dinv)

    const size_t lds_bytes = (size_t)2 * IN_C * OUT_C * sizeof(__bf16); // 256 KB
    static_assert(2 * IN_C * OUT_C * sizeof(__bf16) <= 320 * 1024, "LDS budget");
    (void)hipFuncSetAttribute((const void*)gcn_gemm,
                              hipFuncAttributeMaxDynamicSharedMemorySize,
                              (int)lds_bytes);

    gcn_init_deg <<<(N_NODES + 255) / 256, 256, 0, stream>>>(deg);
    gcn_count_deg<<<(N_EDGES + 255) / 256, 256, 0, stream>>>(ei, deg);
    gcn_rsqrt    <<<(N_NODES + 255) / 256, 256, 0, stream>>>(deg);
    gcn_gemm     <<<(N_NODES + 127) / 128, 256, lds_bytes, stream>>>(x, W, hbuf);
    gcn_init_out <<<(out_size + 255) / 256, 256, 0, stream>>>(hbuf, deg, b, out, out_size);
    gcn_scatter  <<<(N_EDGES + 7) / 8, 256, 0, stream>>>(ei, hbuf, deg, out);
}
